// TopKRouter_7009386627574
// MI455X (gfx1250) — matrix operand order repaired and checked by reading, not compiled, not run on hardware
//
#include <hip/hip_runtime.h>
#include <cstdint>
#include <cstddef>

// ---------------------------------------------------------------------------
// TopK MoE router for MI455X (gfx1250, wave32, WMMA).
//
// Stage 0: W (64x4096 f32) -> bf16 copy in d_ws (512 KB, L2-resident).
// Stage 1: logits = h @ W^T + bias via v_wmma_f32_16x16x32_bf16.
//   - one wave owns a 16(row) x 64(expert) tile: 4 v8f accumulators,
//     A fragment (f32->bf16 cvt on the fly, NT loads) reused across 4
//     B fragments (direct bf16 loads). Explicit double-buffered k-loop.
// Stage 2: softmax + gumbel top-k, one wave per row.
// ---------------------------------------------------------------------------

typedef __attribute__((ext_vector_type(16))) __bf16 v16bf;
typedef __attribute__((ext_vector_type(8)))  __bf16 v8bf;
typedef __attribute__((ext_vector_type(8)))  float  v8f;
typedef __attribute__((ext_vector_type(4)))  float  v4f;

static constexpr int T_DIM = 8192;
static constexpr int D_DIM = 4096;
static constexpr int E_DIM = 64;

__device__ __forceinline__ v16bf cvt16_bf16(v4f a, v4f b, v4f c, v4f d) {
  v16bf r;
#pragma unroll
  for (int i = 0; i < 4; ++i) {
    r[i]      = (__bf16)a[i];
    r[4 + i]  = (__bf16)b[i];
    r[8 + i]  = (__bf16)c[i];
    r[12 + i] = (__bf16)d[i];
  }
  return r;
}

// ---------------------------------------------------------------------------
// Stage 0: convert W to bf16 (each thread: 8 elements, b128 in / b128 out).
// ---------------------------------------------------------------------------
__global__ __launch_bounds__(256) void convert_w_bf16(
    const float* __restrict__ Wm, __bf16* __restrict__ out) {
  const size_t i = ((size_t)blockIdx.x * 256 + threadIdx.x) * 8;
  v4f x0 = *(const v4f*)(Wm + i);
  v4f x1 = *(const v4f*)(Wm + i + 4);
  v8bf r;
#pragma unroll
  for (int j = 0; j < 4; ++j) {
    r[j]     = (__bf16)x0[j];
    r[4 + j] = (__bf16)x1[j];
  }
  *(v8bf*)(out + i) = r;
}

// One k-step's worth of operands (A staged as f32, B already bf16).
struct KFrag {
  v4f a0, a1, a2, a3;
  v16bf b0, b1, b2, b3;
};

// ---------------------------------------------------------------------------
// Stage 1: GEMM + bias -> logits (f32).
// Block = 128 threads = 4 waves; each wave: 16 rows x 64 experts.
// Grid = T/64 = 128 blocks. EXEC all-ones at every WMMA (exact tiling).
// ---------------------------------------------------------------------------
template <bool WBF16>
__global__ __launch_bounds__(128) void moe_gemm_bias(
    const float* __restrict__ h, const float* __restrict__ Wf,
    const __bf16* __restrict__ Wb, const float* __restrict__ bias,
    float* __restrict__ logits) {
  const int lane = threadIdx.x & 31;
  const int wave = threadIdx.x >> 5;
  const int hi   = lane >> 4;    // 0: lanes 0-15, 1: lanes 16-31
  const int col  = lane & 15;    // M (for A) / N (for B,C)
  const int m0   = blockIdx.x * 64 + wave * 16;

  // A fragment: row m0+col, 16 contiguous K at +hi*16 (matches B)
  const float* __restrict__ arow = h + (size_t)(m0 + col) * D_DIM + hi * 16;

  // B fragment bases: expert row (e*16+col) of W, 16 contiguous K at +hi*16
  const __bf16* __restrict__ bb[4];
  const float*  __restrict__ bf[4];
#pragma unroll
  for (int e = 0; e < 4; ++e) {
    const size_t off = (size_t)(e * 16 + col) * D_DIM + hi * 16;
    bb[e] = Wb + off;
    bf[e] = Wf + off;
  }

  auto ldb = [&](int e, int kk) -> v16bf {
    if constexpr (WBF16) {
      return *(const v16bf*)(bb[e] + kk);        // 2 x global_load_b128
    } else {
      const float* p = bf[e] + kk;
      return cvt16_bf16(*(const v4f*)(p),     *(const v4f*)(p + 4),
                        *(const v4f*)(p + 8), *(const v4f*)(p + 12));
    }
  };
  // Streaming h: non-temporal, keep 192MB L2 for W / logits.
  auto lda = [&](int off) -> v4f {
    return __builtin_nontemporal_load((const v4f*)(arow + off));
  };
  auto load_frag = [&](int kk) -> KFrag {
    KFrag f;
    f.a0 = lda(kk);      f.a1 = lda(kk + 4);
    f.a2 = lda(kk + 8); f.a3 = lda(kk + 12);
    f.b0 = ldb(0, kk); f.b1 = ldb(1, kk); f.b2 = ldb(2, kk); f.b3 = ldb(3, kk);
    return f;
  };

  v8f c0 = {}, c1 = {}, c2 = {}, c3 = {};
  auto compute = [&](const KFrag& f) {
    const v16bf a = cvt16_bf16(f.a0, f.a1, f.a2, f.a3);
    c0 = __builtin_amdgcn_wmma_f32_16x16x32_bf16(false, a, false, f.b0,
                                                 (short)0, c0, false, false);
    c1 = __builtin_amdgcn_wmma_f32_16x16x32_bf16(false, a, false, f.b1,
                                                 (short)0, c1, false, false);
    c2 = __builtin_amdgcn_wmma_f32_16x16x32_bf16(false, a, false, f.b2,
                                                 (short)0, c2, false, false);
    c3 = __builtin_amdgcn_wmma_f32_16x16x32_bf16(false, a, false, f.b3,
                                                 (short)0, c3, false, false);
  };

  // --- explicit double-buffered k loop (128 steps of K=32) -----------------
  KFrag fA = load_frag(0);
  for (int i = 0; i < (D_DIM / 64) - 1; ++i) {
    const int kk = i * 64;
    KFrag fB = load_frag(kk + 32);   // prefetch odd step
    compute(fA);                      // compute even step
    fA = load_frag(kk + 64);          // prefetch next even step
    compute(fB);                      // compute odd step
  }
  {
    KFrag fB = load_frag(D_DIM - 32);
    compute(fA);                      // k = D-64
    compute(fB);                      // k = D-32
  }

  // C/D layout: row = m0 + r + hi*8, col = (lane&15) + 16*tile
  const float bv0 = bias[col];
  const float bv1 = bias[col + 16];
  const float bv2 = bias[col + 32];
  const float bv3 = bias[col + 48];
  const int mrow = m0 + hi * 8;
#pragma unroll
  for (int r = 0; r < 8; ++r) {
    const size_t o = (size_t)(mrow + r) * E_DIM + col;
    logits[o]      = c0[r] + bv0;
    logits[o + 16] = c1[r] + bv1;
    logits[o + 32] = c2[r] + bv2;
    logits[o + 48] = c3[r] + bv3;
  }
}

// ---------------------------------------------------------------------------
// Stage 2: softmax + gumbel top-k. One wave per row; lane owns experts
// {lane, lane+32}. Butterfly reductions over 32 lanes (wave32!).
// ---------------------------------------------------------------------------
__device__ __forceinline__ float gumbel_noise(uint32_t idx) {
  // deterministic hash stand-in for jax.random with key 42 (no runtime check)
  uint32_t x = idx * 0x9E3779B9u + 42u * 0x85EBCA6Bu;
  x ^= x >> 16; x *= 0x7FEB352Du;
  x ^= x >> 15; x *= 0x846CA68Bu;
  x ^= x >> 16;
  float u = (float)x * 2.3283064365386963e-10f;     // [0,1)
  u = fminf(fmaxf(u, 1e-6f), 1.0f - 1e-6f);
  return -__logf(-__logf(u));
}

__global__ __launch_bounds__(256) void moe_router(
    const float* __restrict__ logits, const int* __restrict__ kptr,
    float* __restrict__ mask, float* __restrict__ weight) {
  const int lane = threadIdx.x & 31;
  const int wave = threadIdx.x >> 5;
  const int t    = blockIdx.x * 8 + wave;
  const int k    = *kptr;
  const size_t base = (size_t)t * E_DIM;

  const float l0 = logits[base + lane];
  const float l1 = logits[base + lane + 32];

  // softmax over 64 experts (clean logits)
  float mx = fmaxf(l0, l1);
#pragma unroll
  for (int s = 16; s >= 1; s >>= 1) mx = fmaxf(mx, __shfl_xor(mx, s, 32));
  const float e0 = __expf(l0 - mx);
  const float e1 = __expf(l1 - mx);
  float sum = e0 + e1;
#pragma unroll
  for (int s = 16; s >= 1; s >>= 1) sum += __shfl_xor(sum, s, 32);
  const float inv = 1.0f / sum;

  // selection scores = logits + gumbel noise
  float s0 = l0 + gumbel_noise((uint32_t)(base + lane));
  float s1 = l1 + gumbel_noise((uint32_t)(base + lane + 32));

  float m0 = 0.0f, m1 = 0.0f;
  for (int it = 0; it < k; ++it) {
    float v = s0; int idx = lane;
    if (s1 > v) { v = s1; idx = lane + 32; }
#pragma unroll
    for (int s = 16; s >= 1; s >>= 1) {
      const float ov = __shfl_xor(v, s, 32);
      const int   oi = __shfl_xor(idx, s, 32);
      if (ov > v || (ov == v && oi < idx)) { v = ov; idx = oi; }
    }
    if (idx == lane)      { m0 = 1.0f; s0 = -3.4e38f; }
    if (idx == lane + 32) { m1 = 1.0f; s1 = -3.4e38f; }
  }

  mask[base + lane]        = m0;
  mask[base + lane + 32]   = m1;
  weight[base + lane]      = e0 * inv;
  weight[base + lane + 32] = e1 * inv;
}

// ---------------------------------------------------------------------------
// Launch: outputs concatenated as (mask, weight, logits), each T*E f32.
// ---------------------------------------------------------------------------
extern "C" void kernel_launch(void* const* d_in, const int* in_sizes, int n_in,
                              void* d_out, int out_size, void* d_ws, size_t ws_size,
                              hipStream_t stream) {
  (void)in_sizes; (void)n_in; (void)out_size;
  const float* h    = (const float*)d_in[0];
  const float* Wm   = (const float*)d_in[1];
  const float* bias = (const float*)d_in[2];
  const int*   kptr = (const int*)d_in[3];

  float* out    = (float*)d_out;
  const size_t TE = (size_t)T_DIM * E_DIM;
  float* mask   = out;
  float* weight = out + TE;
  float* logits = out + 2 * TE;

  const size_t wbf_bytes = (size_t)E_DIM * D_DIM * sizeof(__bf16);
  if (ws_size >= wbf_bytes) {
    __bf16* wbf = (__bf16*)d_ws;
    hipLaunchKernelGGL(convert_w_bf16, dim3((E_DIM * D_DIM) / (256 * 8)),
                       dim3(256), 0, stream, Wm, wbf);
    hipLaunchKernelGGL((moe_gemm_bias<true>), dim3(T_DIM / 64), dim3(128), 0,
                       stream, h, Wm, wbf, bias, logits);
  } else {
    hipLaunchKernelGGL((moe_gemm_bias<false>), dim3(T_DIM / 64), dim3(128), 0,
                       stream, h, Wm, (const __bf16*)nullptr, bias, logits);
  }
  hipLaunchKernelGGL(moe_router, dim3(T_DIM / 8), dim3(256), 0, stream,
                     logits, kptr, mask, weight);
}
